// Agent_global_gridnet_multiepisode_pretrained_85710367359126
// MI455X (gfx1250) — matrix operand order repaired and checked by reading, not compiled, not run on hardware
//
#include <hip/hip_runtime.h>
#include <hip/hip_bf16.h>
#include <math.h>

typedef __attribute__((ext_vector_type(16))) _Float16 v16h;
typedef __attribute__((ext_vector_type(8)))  _Float16 v8h;
typedef __attribute__((ext_vector_type(8)))  float    v8f;

#define B_   16
#define N_   256
#define D_   240
#define H_   8
#define DH_  30
#define FF_  1024
#define NL_  4
#define L_   4
#define S_   1034           // 2*(L+1+2N)
#define M_   (B_*S_)        // 16544 = 1034*16
#define MT_  (M_/16)
#define NM1_ 517
#define NEG_ (-1.0e30f)
#define SP_  1088           // padded key length for V^T (17 blocks of 64)
#define KH_  256            // padded per-row K for D=240 activations (H*32)

#define SHUF16(x, y) __builtin_shufflevector((x), (y), 0,1,2,3,4,5,6,7,8,9,10,11,12,13,14,15)

// ---------------------------------------------------------------------------
// WMMA GEMM:  C[M,N] = act(Ah[M,KP] @ Wp + bias).  Ah: padded f16 row-major.
// Wp packed: Wp[((k/32)*N + col)*32 + (k%32)].  One wave -> 16x32 C strip.
// Software-pipelined (double-buffered fragments), branchless inner loop.
// ---------------------------------------------------------------------------
template<int KP>
__global__ __launch_bounds__(256)
void gemm_wmma_kernel(const _Float16* __restrict__ Ah,
                      const _Float16* __restrict__ Wp,
                      const float* __restrict__ bias,
                      float* __restrict__ Cf, _Float16* __restrict__ Ch,
                      int Mrows, int N, int act)
{
    const int lane  = threadIdx.x;
    const int col16 = lane & 15;
    const int hi    = lane >> 4;
    const int tile_m = blockIdx.y * 8 + threadIdx.y;
    if (tile_m * 16 >= Mrows) return;
    const int tn2  = blockIdx.x;                  // 32-column strip
    const int col0 = tn2 * 32 + col16;
    const bool has2 = (tn2 * 32 + 16) < N;        // uniform per block
    const int col1 = has2 ? col0 + 16 : col0;     // clamp: keeps loads in-bounds

    const _Float16* ap  = Ah + (size_t)(tile_m * 16 + col16) * KP + (hi ? 16 : 0);
    const _Float16* wp0 = Wp + (size_t)col0 * 32 + (hi ? 16 : 0);
    const _Float16* wp1 = Wp + (size_t)col1 * 32 + (hi ? 16 : 0);
    const size_t wstep = (size_t)N * 32;

    v8f acc0 = {0.f,0.f,0.f,0.f,0.f,0.f,0.f,0.f};
    v8f acc1 = {0.f,0.f,0.f,0.f,0.f,0.f,0.f,0.f};

    // prologue: fragments for kt = 0
    v8h a0c = *(const v8h*)(ap);
    v8h a1c = *(const v8h*)(ap + 8);
    v8h b0c = *(const v8h*)(wp0);
    v8h b1c = *(const v8h*)(wp0 + 8);
    v8h c0c = *(const v8h*)(wp1);
    v8h c1c = *(const v8h*)(wp1 + 8);

    const int nkt = KP >> 5;
    #pragma unroll
    for (int kt = 0; kt < nkt - 1; ++kt) {
        ap  += 32;
        wp0 += wstep;
        wp1 += wstep;
        v8h a0n = *(const v8h*)(ap);
        v8h a1n = *(const v8h*)(ap + 8);
        v8h b0n = *(const v8h*)(wp0);
        v8h b1n = *(const v8h*)(wp0 + 8);
        v8h c0n = *(const v8h*)(wp1);
        v8h c1n = *(const v8h*)(wp1 + 8);
        v16h af = SHUF16(a0c, a1c);
        v16h bf = SHUF16(b0c, b1c);
        v16h bg = SHUF16(c0c, c1c);
        acc0 = __builtin_amdgcn_wmma_f32_16x16x32_f16(false, af, false, bf,
                                                      (short)0, acc0, false, false);
        acc1 = __builtin_amdgcn_wmma_f32_16x16x32_f16(false, af, false, bg,
                                                      (short)0, acc1, false, false);
        a0c = a0n; a1c = a1n; b0c = b0n; b1c = b1n; c0c = c0n; c1c = c1n;
    }
    {
        v16h af = SHUF16(a0c, a1c);
        v16h bf = SHUF16(b0c, b1c);
        v16h bg = SHUF16(c0c, c1c);
        acc0 = __builtin_amdgcn_wmma_f32_16x16x32_f16(false, af, false, bf,
                                                      (short)0, acc0, false, false);
        acc1 = __builtin_amdgcn_wmma_f32_16x16x32_f16(false, af, false, bg,
                                                      (short)0, acc1, false, false);
    }

    const float bv0 = bias[col0];
    const float bv1 = has2 ? bias[col0 + 16] : 0.f;
    #pragma unroll
    for (int r = 0; r < 8; ++r) {
        int orow = tile_m * 16 + r + (hi ? 8 : 0);
        float v = acc0[r] + bv0;
        if (act) v = 0.5f * v * (1.f + erff(v * 0.70710678118654752f));
        if (Ch) Ch[(size_t)orow * N + col0] = (_Float16)v;
        else    Cf[(size_t)orow * N + col0] = v;
        if (has2) {
            float u = acc1[r] + bv1;
            if (act) u = 0.5f * u * (1.f + erff(u * 0.70710678118654752f));
            if (Ch) Ch[(size_t)orow * N + col0 + 16] = (_Float16)u;
            else    Cf[(size_t)orow * N + col0 + 16] = u;
        }
    }
}

// ---------------------------------------------------------------------------
// Weight packing. mode 0: row c -> c (c<K else 0). mode 1: c=h*32+d -> h*30+d.
// ---------------------------------------------------------------------------
__global__ void pack_w_kernel(const float* __restrict__ W, _Float16* __restrict__ Wp,
                              int K, int Kp, int N, int mode)
{
    int idx = blockIdx.x * blockDim.x + threadIdx.x;
    if (idx >= Kp * N) return;
    int c   = idx / N;
    int col = idx % N;
    float v = 0.f;
    if (mode == 0) {
        if (c < K) v = W[(size_t)c * N + col];
    } else {
        int h = c >> 5, d = c & 31;
        if (d < DH_) v = W[(size_t)(h * DH_ + d) * N + col];
    }
    int kt = c >> 5, kk = c & 31;
    Wp[((size_t)kt * N + col) * 32 + kk] = (_Float16)v;
}

// ---------------------------------------------------------------------------
// Fused RoPE + cast + scale: fp32 [M,240] -> f16 [M,H*32] interleaved, pad=0.
// `mul` folds 1/sqrt(DH) into Q.
// ---------------------------------------------------------------------------
__global__ void rope_cast_kernel(const float* __restrict__ X,
                                 const float* __restrict__ coords,
                                 _Float16* __restrict__ Xh, float mul)
{
    int idx = blockIdx.x * blockDim.x + threadIdx.x;
    if (idx >= M_ * H_ * 16) return;
    int t   = idx & 15;
    int h   = (idx >> 4) & 7;
    int row = idx >> 7;
    _Float16* orow = Xh + (size_t)row * KH_ + h * 32;
    if (t == 15) { orow[30] = (_Float16)0.f; orow[31] = (_Float16)0.f; return; }
    int a = t / 5, j = t % 5;
    int d1 = a * 10 + j;
    float invf = __expf(-(float)j * 1.8420680743952368f);  // 10000^(-j/5)
    float ang  = coords[(size_t)row * 3 + a] * invf;
    float c = __cosf(ang), s = __sinf(ang);
    float x1 = X[(size_t)row * D_ + h * DH_ + d1];
    float x2 = X[(size_t)row * D_ + h * DH_ + d1 + 5];
    orow[d1]     = (_Float16)((x1 * c - x2 * s) * mul);
    orow[d1 + 5] = (_Float16)((x1 * s + x2 * c) * mul);
}

// V fp32 [M,240] -> f16 transposed [B,H,32,SP_].  Row d==30 is all-ones so the
// P@V WMMA also accumulates the softmax denominator; d==31 and s>=S are zero.
__global__ void cast_vt_kernel(const float* __restrict__ V, _Float16* __restrict__ Vt)
{
    int idx = blockIdx.x * blockDim.x + threadIdx.x;
    if (idx >= B_ * H_ * 32 * SP_) return;
    int s  = idx % SP_;
    int r  = idx / SP_;
    int d  = r & 31;
    int h  = (r >> 5) & 7;
    int b  = r >> 8;
    float v;
    if (d == DH_)                 v = 1.0f;             // denominator column
    else if (d < DH_ && s < S_)   v = V[(size_t)(b * S_ + s) * D_ + h * DH_ + d];
    else                          v = 0.f;
    Vt[idx] = (_Float16)v;
}

// ---------------------------------------------------------------------------
// Flash attention, 64 keys per iteration.  One wave per (b,h,16-query tile).
// Row-sum folded into the P@V WMMA (ones column in V^T); only the row-max
// needs a shuffle chain, amortized over 64 keys.
// ---------------------------------------------------------------------------
__global__ __launch_bounds__(256)
void attn_kernel(const _Float16* __restrict__ Qh, const _Float16* __restrict__ Kh,
                 const _Float16* __restrict__ Vt, _Float16* __restrict__ Oh)
{
    __shared__ _Float16 plds[8 * 1024];   // per-wave 16x64 f16 P staging
    const int lane  = threadIdx.x;
    const int col16 = lane & 15;
    const int hi    = lane >> 4;
    const int w     = threadIdx.y;
    const int qt    = blockIdx.x * 8 + w;
    if (qt >= 65) return;
    const int h    = blockIdx.y;
    const int b    = blockIdx.z;

    // Q A-fragment (pre-scaled by 1/sqrt(DH) in rope_cast)
    const int qrow = qt * 16 + col16;
    const int qrc  = qrow < S_ ? qrow : S_ - 1;
    const _Float16* qp = Qh + (size_t)(b * S_ + qrc) * KH_ + h * 32;
    v8h q0 = *(const v8h*)(qp + (hi ? 16 : 0));
    v8h q1 = *(const v8h*)(qp + 8 + (hi ? 16 : 0));
    v16h qa = SHUF16(q0, q1);

    float mrow[8];
    v8f o0 = {0.f,0.f,0.f,0.f,0.f,0.f,0.f,0.f};
    v8f o1 = {0.f,0.f,0.f,0.f,0.f,0.f,0.f,0.f};
    #pragma unroll
    for (int r = 0; r < 8; ++r) mrow[r] = NEG_;

    _Float16* pw = &plds[w * 1024];
    const _Float16* vbase = Vt + (size_t)(b * H_ + h) * 32 * SP_;

    for (int kt4 = 0; kt4 < 17; ++kt4) {       // 64 keys / iteration
        const int kbase = kt4 * 64;
        // ---- issue all global loads up front: 4 K-frags, then 4 V-frags ----
        v16h kf[4];
        #pragma unroll
        for (int jj = 0; jj < 4; ++jj) {
            int key  = kbase + jj * 16 + col16;
            int keyc = key < S_ ? key : S_ - 1;
            const _Float16* kp = Kh + (size_t)(b * S_ + keyc) * KH_ + h * 32 + (hi ? 16 : 0);
            v8h k0 = *(const v8h*)(kp);
            v8h k1 = *(const v8h*)(kp + 8);
            kf[jj] = SHUF16(k0, k1);
        }
        v16h vf[2][2];
        #pragma unroll
        for (int t = 0; t < 2; ++t) {
            const _Float16* vp = vbase + (size_t)(t * 16 + col16) * SP_ + kbase + (hi ? 16 : 0);
            #pragma unroll
            for (int ks = 0; ks < 2; ++ks) {
                v8h v0 = *(const v8h*)(vp + ks * 32);
                v8h v1 = *(const v8h*)(vp + ks * 32 + 8);
                vf[t][ks] = SHUF16(v0, v1);
            }
        }

        // ---- scores (4 tiles) + mask ----
        v8f sc[4];
        #pragma unroll
        for (int jj = 0; jj < 4; ++jj) {
            v8f z = {0.f,0.f,0.f,0.f,0.f,0.f,0.f,0.f};
            z = __builtin_amdgcn_wmma_f32_16x16x32_f16(false, qa, false, kf[jj],
                                                       (short)0, z, false, false);
            int key = kbase + jj * 16 + col16;
            #pragma unroll
            for (int r = 0; r < 8; ++r) {
                int qr = qt * 16 + r + (hi ? 8 : 0);
                float sv = z[r];
                if (key >= S_ || (qr < NM1_ && key >= NM1_)) sv = NEG_;
                z[r] = sv;
            }
            sc[jj] = z;
        }

        // ---- online max (one shuffle chain per row per 64 keys) ----
        float corr[8];
        #pragma unroll
        for (int r = 0; r < 8; ++r) {
            float mv = fmaxf(fmaxf(sc[0][r], sc[1][r]), fmaxf(sc[2][r], sc[3][r]));
            mv = fmaxf(mv, __shfl_xor(mv, 1, 32));
            mv = fmaxf(mv, __shfl_xor(mv, 2, 32));
            mv = fmaxf(mv, __shfl_xor(mv, 4, 32));
            mv = fmaxf(mv, __shfl_xor(mv, 8, 32));
            float mnew = fmaxf(mrow[r], mv);
            corr[r] = __expf(mrow[r] - mnew);
            mrow[r] = mnew;
        }
        #pragma unroll
        for (int jj = 0; jj < 4; ++jj)
            #pragma unroll
            for (int r = 0; r < 8; ++r)
                sc[jj][r] = __expf(sc[jj][r] - mrow[r]);
        #pragma unroll
        for (int r = 0; r < 8; ++r) { o0[r] *= corr[r]; o1[r] *= corr[r]; }

        // ---- P (C-layout) -> A-fragments via per-wave LDS (16x64) ----
        #pragma unroll
        for (int jj = 0; jj < 4; ++jj)
            #pragma unroll
            for (int r = 0; r < 8; ++r)
                pw[(r + (hi ? 8 : 0)) * 64 + jj * 16 + col16] = (_Float16)sc[jj][r];
        asm volatile("s_wait_dscnt 0" ::: "memory");
        v16h pa[2];
        #pragma unroll
        for (int ks = 0; ks < 2; ++ks) {
            v8h pA = *(const v8h*)(pw + col16 * 64 + ks * 32 + (hi ? 16 : 0));
            v8h pB = *(const v8h*)(pw + col16 * 64 + ks * 32 + 8 + (hi ? 16 : 0));
            pa[ks] = SHUF16(pA, pB);
        }

        // ---- O += P @ V (also accumulates row-sum in d==30 column) ----
        #pragma unroll
        for (int ks = 0; ks < 2; ++ks) {
            o0 = __builtin_amdgcn_wmma_f32_16x16x32_f16(false, pa[ks], false, vf[0][ks],
                                                        (short)0, o0, false, false);
            o1 = __builtin_amdgcn_wmma_f32_16x16x32_f16(false, pa[ks], false, vf[1][ks],
                                                        (short)0, o1, false, false);
        }
    }

    // ---- epilogue: broadcast denominator (o1 column d==30 -> lane hi*16+14) ----
    #pragma unroll
    for (int r = 0; r < 8; ++r) {
        int qr = qt * 16 + r + (hi ? 8 : 0);
        float l = __shfl(o1[r], (lane & 16) + 14, 32);
        if (qr < S_) {
            float inv = 1.f / l;
            _Float16* orow = Oh + (size_t)(b * S_ + qr) * KH_ + h * 32;
            orow[col16] = (_Float16)(o0[r] * inv);
            int d1 = 16 + col16;
            float u = (d1 < DH_) ? o1[r] * inv : 0.f;   // zero the pad columns
            orow[d1] = (_Float16)u;
        }
    }
}

// ---------------------------------------------------------------------------
// Build concatenated src: fp32 [M,240] + padded f16 [M,256]
// ---------------------------------------------------------------------------
__global__ void build_src_kernel(const float* __restrict__ ht_t, const float* __restrict__ hd_t,
                                 const float* __restrict__ ht_m1, const float* __restrict__ hd_m1,
                                 const float* __restrict__ st_t, const float* __restrict__ st_m1,
                                 const float* __restrict__ tok_m1, const float* __restrict__ tok_t,
                                 float* __restrict__ src, _Float16* __restrict__ xh)
{
    int idx = blockIdx.x * blockDim.x + threadIdx.x;
    if (idx >= M_ * KH_) return;
    int d   = idx % KH_;
    int row = idx / KH_;
    if (d >= D_) { xh[idx] = (_Float16)0.f; return; }
    int b = row / S_;
    int p = row % S_;
    const float* sp;
    if      (p == 0)    sp = st_m1  + (size_t)b * D_;
    else if (p < 257)   sp = ht_m1  + ((size_t)b * N_ + (p - 1))   * D_;
    else if (p < 513)   sp = hd_m1  + ((size_t)b * N_ + (p - 257)) * D_;
    else if (p < 517)   sp = tok_m1 + (size_t)(p - 513) * D_;
    else if (p == 517)  sp = st_t   + (size_t)b * D_;
    else if (p < 774)   sp = ht_t   + ((size_t)b * N_ + (p - 518)) * D_;
    else if (p < 1030)  sp = hd_t   + ((size_t)b * N_ + (p - 774)) * D_;
    else                sp = tok_t  + (size_t)(p - 1030) * D_;
    float v = sp[d];
    src[(size_t)row * D_ + d] = v;
    xh[idx] = (_Float16)v;
}

__global__ void build_coords_kernel(const float* __restrict__ ch_t, const float* __restrict__ cd_t,
                                    const float* __restrict__ ch_m1, const float* __restrict__ cd_m1,
                                    const float* __restrict__ tr_t, const float* __restrict__ tr_m1,
                                    float* __restrict__ coords)
{
    int idx = blockIdx.x * blockDim.x + threadIdx.x;
    if (idx >= M_ * 3) return;
    int a   = idx % 3;
    int row = idx / 3;
    int b   = row / S_;
    int p   = row % S_;
    const float* sp;
    if      (p == 0)    sp = tr_m1 + (size_t)b * 3;
    else if (p < 257)   sp = ch_m1 + ((size_t)b * N_ + (p - 1))   * 3;
    else if (p < 513)   sp = cd_m1 + ((size_t)b * N_ + (p - 257)) * 3;
    else if (p < 517)   sp = tr_m1 + (size_t)b * 3;
    else if (p == 517)  sp = tr_t  + (size_t)b * 3;
    else if (p < 774)   sp = ch_t  + ((size_t)b * N_ + (p - 518)) * 3;
    else if (p < 1030)  sp = cd_t  + ((size_t)b * N_ + (p - 774)) * 3;
    else                sp = tr_t  + (size_t)b * 3;
    coords[idx] = sp[a];
}

// ---------------------------------------------------------------------------
// Y = LayerNorm(X + Dlt)*g + be ; fp32 out + fused padded f16 out.
// ---------------------------------------------------------------------------
__global__ __launch_bounds__(256)
void add_ln_kernel(const float* __restrict__ X, const float* __restrict__ Dlt,
                   const float* __restrict__ g, const float* __restrict__ be,
                   float* __restrict__ Y, _Float16* __restrict__ Yh, int Mrows)
{
    int row  = blockIdx.x * 8 + (threadIdx.x >> 5);
    if (row >= Mrows) return;
    int lane = threadIdx.x & 31;
    float vals[8];
    float sum = 0.f;
    #pragma unroll
    for (int i = 0; i < 8; ++i) {
        int c = lane + i * 32;
        float v = 0.f;
        if (c < D_) v = X[(size_t)row * D_ + c] + Dlt[(size_t)row * D_ + c];
        vals[i] = v;
        sum += v;
    }
    sum += __shfl_xor(sum, 1, 32);  sum += __shfl_xor(sum, 2, 32);
    sum += __shfl_xor(sum, 4, 32);  sum += __shfl_xor(sum, 8, 32);
    sum += __shfl_xor(sum, 16, 32);
    float mu = sum / (float)D_;
    float vs = 0.f;
    #pragma unroll
    for (int i = 0; i < 8; ++i) {
        int c = lane + i * 32;
        if (c < D_) { float d = vals[i] - mu; vs += d * d; }
    }
    vs += __shfl_xor(vs, 1, 32);  vs += __shfl_xor(vs, 2, 32);
    vs += __shfl_xor(vs, 4, 32);  vs += __shfl_xor(vs, 8, 32);
    vs += __shfl_xor(vs, 16, 32);
    float rs = rsqrtf(vs / (float)D_ + 1e-5f);
    #pragma unroll
    for (int i = 0; i < 8; ++i) {
        int c = lane + i * 32;
        if (c < D_) {
            float o = (vals[i] - mu) * rs * g[c] + be[c];
            Y[(size_t)row * D_ + c] = o;
            Yh[(size_t)row * KH_ + c] = (_Float16)o;
        } else {
            Yh[(size_t)row * KH_ + c] = (_Float16)0.f;
        }
    }
}

__global__ void extract_kernel(const float* __restrict__ src, float* __restrict__ out)
{
    int idx = blockIdx.x * blockDim.x + threadIdx.x;
    if (idx >= B_ * L_ * D_) return;
    int d = idx % D_;
    int l = (idx / D_) % L_;
    int b = idx / (D_ * L_);
    out[idx] = src[(size_t)(b * S_ + (S_ - L_) + l) * D_ + d];
}

// ---------------------------------------------------------------------------
extern "C" void kernel_launch(void* const* d_in, const int* in_sizes, int n_in,
                              void* d_out, int out_size, void* d_ws, size_t ws_size,
                              hipStream_t stream)
{
    (void)in_sizes; (void)n_in; (void)out_size; (void)ws_size;
    const float* ht_t  = (const float*)d_in[0];
    const float* hd_t  = (const float*)d_in[1];
    const float* ht_m1 = (const float*)d_in[2];
    const float* hd_m1 = (const float*)d_in[3];
    const float* ch_t  = (const float*)d_in[4];
    const float* cd_t  = (const float*)d_in[5];
    const float* ch_m1 = (const float*)d_in[6];
    const float* cd_m1 = (const float*)d_in[7];
    const float* st_t  = (const float*)d_in[8];
    const float* st_m1 = (const float*)d_in[9];
    const float* tr_t  = (const float*)d_in[10];
    const float* tr_m1 = (const float*)d_in[11];
    const float* tokm1 = (const float*)d_in[12];
    const float* tokt  = (const float*)d_in[13];

    float* ws = (float*)d_ws;
    const size_t MD = (size_t)M_ * D_;
    float* src    = ws;
    float* src2   = src  + MD;
    float* qb     = src2 + MD;   // Q fp32, later Wo-proj fp32
    float* kb     = qb   + MD;   // K fp32
    float* vb     = kb   + MD;   // V fp32, later FF2 output fp32
    float* coords = vb   + MD;
    size_t nfl = 5 * MD + (size_t)M_ * 3;
    nfl = (nfl + 3) & ~(size_t)3;            // 16B-align the f16 region
    _Float16* hb   = (_Float16*)(ws + nfl);
    const size_t M256 = (size_t)M_ * KH_;
    _Float16* xh   = hb;                     // padded activations (src / src2)
    _Float16* qh   = xh   + M256;
    _Float16* kh   = qh   + M256;
    _Float16* abh  = kh   + M256;            // attention output (head-interleaved)
    _Float16* ffhh = abh  + M256;            // FF1 output f16 [M,1024]
    _Float16* vt   = ffhh + (size_t)M_ * FF_;
    _Float16* wp   = vt   + (size_t)B_ * H_ * 32 * SP_;   // packed weights

    dim3 blk(256);
    dim3 blk2(32, 8);
    dim3 gN240((D_ + 31) / 32, (MT_ + 7) / 8);
    dim3 gNff((FF_ + 31) / 32, (MT_ + 7) / 8);
    dim3 gattn(9, H_, B_);
    const float qscale = 0.18257418583505536f;  // 1/sqrt(30)

    build_src_kernel<<<(M_ * KH_ + 255) / 256, blk, 0, stream>>>(
        ht_t, hd_t, ht_m1, hd_m1, st_t, st_m1, tokm1, tokt, src, xh);
    build_coords_kernel<<<(M_ * 3 + 255) / 256, blk, 0, stream>>>(
        ch_t, cd_t, ch_m1, cd_m1, tr_t, tr_m1, coords);

    for (int i = 0; i < NL_; ++i) {
        const float* Wq  = (const float*)d_in[14] + (size_t)i * D_ * D_;
        const float* bq  = (const float*)d_in[15] + (size_t)i * D_;
        const float* Wk  = (const float*)d_in[16] + (size_t)i * D_ * D_;
        const float* bk  = (const float*)d_in[17] + (size_t)i * D_;
        const float* Wv  = (const float*)d_in[18] + (size_t)i * D_ * D_;
        const float* bv  = (const float*)d_in[19] + (size_t)i * D_;
        const float* Wo  = (const float*)d_in[20] + (size_t)i * D_ * D_;
        const float* bo  = (const float*)d_in[21] + (size_t)i * D_;
        const float* W1  = (const float*)d_in[22] + (size_t)i * D_ * FF_;
        const float* b1  = (const float*)d_in[23] + (size_t)i * FF_;
        const float* W2  = (const float*)d_in[24] + (size_t)i * FF_ * D_;
        const float* b2  = (const float*)d_in[25] + (size_t)i * D_;
        const float* g1  = (const float*)d_in[26] + (size_t)i * D_;
        const float* be1 = (const float*)d_in[27] + (size_t)i * D_;
        const float* g2  = (const float*)d_in[28] + (size_t)i * D_;
        const float* be2 = (const float*)d_in[29] + (size_t)i * D_;

        const int pk240 = (KH_ * D_ + 255) / 256;
        const int pkff  = (KH_ * FF_ + 255) / 256;
        const int pkw2  = (FF_ * D_ + 255) / 256;

        // QKV
        pack_w_kernel<<<pk240, blk, 0, stream>>>(Wq, wp, D_, KH_, D_, 0);
        gemm_wmma_kernel<KH_><<<gN240, blk2, 0, stream>>>(xh, wp, bq, qb, nullptr, M_, D_, 0);
        pack_w_kernel<<<pk240, blk, 0, stream>>>(Wk, wp, D_, KH_, D_, 0);
        gemm_wmma_kernel<KH_><<<gN240, blk2, 0, stream>>>(xh, wp, bk, kb, nullptr, M_, D_, 0);
        pack_w_kernel<<<pk240, blk, 0, stream>>>(Wv, wp, D_, KH_, D_, 0);
        gemm_wmma_kernel<KH_><<<gN240, blk2, 0, stream>>>(xh, wp, bv, vb, nullptr, M_, D_, 0);

        // RoPE + layout conversion (Q gets 1/sqrt(DH) folded in)
        rope_cast_kernel<<<(M_ * H_ * 16 + 255) / 256, blk, 0, stream>>>(qb, coords, qh, qscale);
        rope_cast_kernel<<<(M_ * H_ * 16 + 255) / 256, blk, 0, stream>>>(kb, coords, kh, 1.0f);
        cast_vt_kernel<<<(B_ * H_ * 32 * SP_ + 255) / 256, blk, 0, stream>>>(vb, vt);

        // Attention
        attn_kernel<<<gattn, blk2, 0, stream>>>(qh, kh, vt, abh);

        // Wo projection (head-interleaved packed weights)
        pack_w_kernel<<<pk240, blk, 0, stream>>>(Wo, wp, D_, KH_, D_, 1);
        gemm_wmma_kernel<KH_><<<gN240, blk2, 0, stream>>>(abh, wp, bo, qb, nullptr, M_, D_, 0);
        add_ln_kernel<<<(M_ + 7) / 8, blk, 0, stream>>>(src, qb, g1, be1, src2, xh, M_);

        // FFN: FF1 (GELU, f16 out), FF2 (fp32 out)
        pack_w_kernel<<<pkff, blk, 0, stream>>>(W1, wp, D_, KH_, FF_, 0);
        gemm_wmma_kernel<KH_><<<gNff, blk2, 0, stream>>>(xh, wp, b1, nullptr, ffhh, M_, FF_, 1);
        pack_w_kernel<<<pkw2, blk, 0, stream>>>(W2, wp, FF_, FF_, D_, 0);
        gemm_wmma_kernel<FF_><<<gN240, blk2, 0, stream>>>(ffhh, wp, b2, vb, nullptr, M_, D_, 0);
        add_ln_kernel<<<(M_ + 7) / 8, blk, 0, stream>>>(src2, vb, g2, be2, src, xh, M_);
    }

    extract_kernel<<<(B_ * L_ * D_ + 255) / 256, blk, 0, stream>>>(src, (float*)d_out);
}
